// Qwen3MoeSparseMoeBlock_88630945120691
// MI455X (gfx1250) — matrix operand order repaired and checked by reading, not compiled, not run on hardware
//
#include <hip/hip_runtime.h>

// Problem dims (compile-time constants from the reference)
#define T_TOK 2048      // B*S
#define H_DIM 2048
#define I_DIM 768
#define E_NUM 64
#define K_TOP 8
#define TK    (T_TOK * K_TOP)   // 16384 routed rows

typedef unsigned short ushort_t;
typedef __attribute__((ext_vector_type(16))) __bf16 v16bf;
typedef __attribute__((ext_vector_type(8)))  float  v8f;
typedef __attribute__((ext_vector_type(4)))  int    v4i;

union FragAB { v16bf v; uint4 q[2]; };

// fp32 -> bf16, round-half-up (1 ulp tie behavior vs RNE; cheap: 2 VALU)
__device__ __forceinline__ ushort_t f2bf(float f) {
  return (ushort_t)((__float_as_uint(f) + 0x8000u) >> 16);
}
__device__ __forceinline__ float bf2f(ushort_t h) {
  return __uint_as_float(((unsigned)h) << 16);
}
// pack two fp32 -> packed bf16x2: one v_cvt_pk_bf16_f32 if available,
// else 2x v_add + 1x v_perm_b32 (select high halves of rounded bits)
__device__ __forceinline__ unsigned pkbf(float a, float b) {
#if __has_builtin(__builtin_amdgcn_cvt_pk_bf16_f32)
  typedef __attribute__((ext_vector_type(2))) __bf16 v2bf;
  union { v2bf v; unsigned u; } cv;
  cv.v = __builtin_amdgcn_cvt_pk_bf16_f32(a, b);
  return cv.u;
#else
  unsigned pa = __float_as_uint(a) + 0x8000u;
  unsigned pb = __float_as_uint(b) + 0x8000u;
  // result bytes [3:0] = {pb.b3, pb.b2, pa.b3, pa.b2}; S0 bytes are 4..7, S1 bytes 0..3
  return __builtin_amdgcn_perm(pb, pa, 0x07060302u);
#endif
}

// 16B global(bf16 data) -> LDS, async on CDNA5 when the builtin exists
#if __has_builtin(__builtin_amdgcn_global_load_async_to_lds_b128)
typedef __attribute__((address_space(1))) v4i* gas_v4i_ptr;   // global (AS1)
typedef __attribute__((address_space(3))) v4i* las_v4i_ptr;   // LDS (AS3)
#endif

__device__ __forceinline__ void copy16_g2l(const ushort_t* __restrict__ g, ushort_t* l) {
#if __has_builtin(__builtin_amdgcn_global_load_async_to_lds_b128)
  __builtin_amdgcn_global_load_async_to_lds_b128((gas_v4i_ptr)g, (las_v4i_ptr)l, 0, 0);
#else
  *(uint4*)l = *(const uint4*)g;
#endif
}
__device__ __forceinline__ void wait_async_copies() {
#if __has_builtin(__builtin_amdgcn_global_load_async_to_lds_b128)
#if __has_builtin(__builtin_amdgcn_s_wait_asynccnt)
  __builtin_amdgcn_s_wait_asynccnt(0);
#else
  asm volatile("s_wait_asynccnt 0x0" ::: "memory");
#endif
#endif
}

// ---------------------------------------------------------------- x -> bf16
__global__ void cvt_x_kernel(const float* __restrict__ x, ushort_t* __restrict__ xb, int n4) {
  int i = blockIdx.x * blockDim.x + threadIdx.x;
  if (i < n4) {
    float4 f = ((const float4*)x)[i];
    uint2 p; p.x = pkbf(f.x, f.y); p.y = pkbf(f.z, f.w);
    ((uint2*)xb)[i] = p;
  }
}

// ---------------------------------------------------------------- router: logits + top8 + softmax
__global__ __launch_bounds__(64) void router_kernel(const float* __restrict__ x,
                                                    const float* __restrict__ gw,
                                                    float* __restrict__ rw,
                                                    int* __restrict__ sel) {
  int t = blockIdx.x;
  int e = threadIdx.x;                 // 64 threads = 64 experts
  __shared__ float xs[256];
  __shared__ float logits[E_NUM];
  float acc = 0.f;
  for (int c = 0; c < H_DIM; c += 256) {
    ((float4*)xs)[e] = ((const float4*)(x + (size_t)t * H_DIM + c))[e];
    __syncthreads();
    #pragma unroll 8
    for (int j = 0; j < 256; ++j) acc += xs[j] * gw[(size_t)(c + j) * E_NUM + e];
    __syncthreads();
  }
  logits[e] = acc;
  __syncthreads();
  if (e == 0) {
    float w[K_TOP]; int idx[K_TOP];
    for (int k = 0; k < K_TOP; ++k) {
      int bi = 0; float bv = logits[0];
      for (int i = 1; i < E_NUM; ++i) if (logits[i] > bv) { bv = logits[i]; bi = i; }
      w[k] = bv; idx[k] = bi; logits[bi] = -3.4e38f;
    }
    float mx = w[0], s = 0.f, ex[K_TOP];
    for (int k = 0; k < K_TOP; ++k) { ex[k] = __expf(w[k] - mx); s += ex[k]; }
    float inv = 1.f / s;
    for (int k = 0; k < K_TOP; ++k) { rw[t * K_TOP + k] = ex[k] * inv; sel[t * K_TOP + k] = idx[k]; }
  }
}

// ---------------------------------------------------------------- per-expert counts
__global__ void count_kernel(const int* __restrict__ sel, int* __restrict__ counts) {
  int i = blockIdx.x * blockDim.x + threadIdx.x;
  if (i < TK) atomicAdd(&counts[sel[i]], 1);
}

// ---------------------------------------------------------------- offsets, tile prefixes, stable lists
__global__ __launch_bounds__(64) void build_kernel(const int* __restrict__ sel,
                                                   const int* __restrict__ counts,
                                                   int* __restrict__ offs,
                                                   int* __restrict__ tgu, int* __restrict__ tdn,
                                                   int* __restrict__ row_tk, int* __restrict__ inv_row) {
  __shared__ int so[E_NUM + 1];
  int e = threadIdx.x;
  if (e == 0) {
    int a = 0, tg = 0, td = 0;
    for (int i = 0; i < E_NUM; ++i) {
      so[i] = a; offs[i] = a; tgu[i] = tg; tdn[i] = td;
      int c = counts[i];
      int mt = (c + 63) >> 6;
      a += c; tg += mt * (I_DIM / 64); td += mt * (H_DIM / 64);
    }
    so[E_NUM] = a; offs[E_NUM] = a; tgu[E_NUM] = tg; tdn[E_NUM] = td;
  }
  __syncthreads();
  int pos = so[e];                         // stable: token order within each expert
  for (int i = 0; i < TK; ++i) {
    if (sel[i] == e) { row_tk[pos] = i; inv_row[i] = pos; ++pos; }
  }
}

// binary search: last e with pfx[e] <= tid
__device__ __forceinline__ int find_expert(const int* __restrict__ pfx, int tid) {
  int lo = 0, hi = E_NUM;
  while (hi - lo > 1) { int mid = (lo + hi) >> 1; if (pfx[mid] <= tid) lo = mid; else hi = mid; }
  return lo;
}

// ---------------------------------------------------------------- grouped GEMM: gate & up + SiLU fuse
// 64x64 tile per block (256 threads = 8 waves; wave = 16 rows x 32 cols), K-step = 32.
__global__ __launch_bounds__(256) void moe_gateup_kernel(
    const ushort_t* __restrict__ xb, const float* __restrict__ gpw, const float* __restrict__ upw,
    const int* __restrict__ offs, const int* __restrict__ tgu, const int* __restrict__ row_tk,
    ushort_t* __restrict__ act) {
  int tid = blockIdx.x;
  if (tid >= tgu[E_NUM]) return;
  int e = find_expert(tgu, tid);
  int cnt = offs[e + 1] - offs[e];
  int mtiles = (cnt + 63) >> 6;
  int local = tid - tgu[e];
  int mt = local % mtiles;
  int nt = local / mtiles;

  __shared__ __align__(16) ushort_t As[64][32];
  __shared__ __align__(16) ushort_t Bg[64][32];   // transposed: [n][k]
  __shared__ __align__(16) ushort_t Bu[64][32];
  __shared__ int rowTok[64];

  int tx = threadIdx.x;
  if (tx < 64) {
    int r = mt * 64 + tx;
    rowTok[tx] = (r < cnt) ? (row_tk[offs[e] + r] / K_TOP) : 0;  // clamp: garbage rows masked at store
  }
  __syncthreads();

  // hoisted per-thread addressing
  int ar = tx >> 2, aseg = tx & 3;                       // A: 64 rows x (4 x 16B)
  const ushort_t* aSrc = xb + (size_t)rowTok[ar] * H_DIM + aseg * 8;
  ushort_t* aDst = &As[ar][aseg * 8];
  int bn = tx & 63, bkh = tx >> 6;                       // B: column n, 8 consecutive k
  size_t bBase = ((size_t)e * H_DIM + bkh * 8) * I_DIM + nt * 64 + bn;

  int wid = tx >> 5, lane = tx & 31;
  int wm = wid & 3, wn = wid >> 2;
  v8f cg0 = {}, cg1 = {}, cu0 = {}, cu1 = {};

  for (int kc = 0; kc < H_DIM / 32; ++kc) {
    copy16_g2l(aSrc + kc * 32, aDst);                    // A tile (bf16, async -> LDS)
    {
      const float* gp = gpw + bBase + (size_t)kc * 32 * I_DIM;
      const float* up = upw + bBase + (size_t)kc * 32 * I_DIM;
      if (kc + 1 < H_DIM / 32) {
        __builtin_prefetch(gp + (size_t)32 * I_DIM, 0, 1);
        __builtin_prefetch(up + (size_t)32 * I_DIM, 0, 1);
      }
      float g[8], u[8];
      #pragma unroll
      for (int i = 0; i < 8; ++i) { g[i] = gp[(size_t)i * I_DIM]; u[i] = up[(size_t)i * I_DIM]; }
      uint4 pg, pu;
      pg.x = pkbf(g[0], g[1]); pg.y = pkbf(g[2], g[3]);
      pg.z = pkbf(g[4], g[5]); pg.w = pkbf(g[6], g[7]);
      pu.x = pkbf(u[0], u[1]); pu.y = pkbf(u[2], u[3]);
      pu.z = pkbf(u[4], u[5]); pu.w = pkbf(u[6], u[7]);
      *(uint4*)&Bg[bn][bkh * 8] = pg;
      *(uint4*)&Bu[bn][bkh * 8] = pu;
    }
    wait_async_copies();
    __syncthreads();

    FragAB a;
    { int row = wm * 16 + (lane & 15); int kb = (lane < 16) ? 0 : 16;
      a.q[0] = *(const uint4*)&As[row][kb];
      a.q[1] = *(const uint4*)&As[row][kb + 8]; }
    FragAB bg0, bg1, bu0, bu1;
    { int kb = (lane < 16) ? 0 : 16;
      int c0 = wn * 32 + (lane & 15);
      bg0.q[0] = *(const uint4*)&Bg[c0][kb];      bg0.q[1] = *(const uint4*)&Bg[c0][kb + 8];
      bg1.q[0] = *(const uint4*)&Bg[c0 + 16][kb]; bg1.q[1] = *(const uint4*)&Bg[c0 + 16][kb + 8];
      bu0.q[0] = *(const uint4*)&Bu[c0][kb];      bu0.q[1] = *(const uint4*)&Bu[c0][kb + 8];
      bu1.q[0] = *(const uint4*)&Bu[c0 + 16][kb]; bu1.q[1] = *(const uint4*)&Bu[c0 + 16][kb + 8]; }

    cg0 = __builtin_amdgcn_wmma_f32_16x16x32_bf16(false, a.v, false, bg0.v, (short)0, cg0, false, false);
    cg1 = __builtin_amdgcn_wmma_f32_16x16x32_bf16(false, a.v, false, bg1.v, (short)0, cg1, false, false);
    cu0 = __builtin_amdgcn_wmma_f32_16x16x32_bf16(false, a.v, false, bu0.v, (short)0, cu0, false, false);
    cu1 = __builtin_amdgcn_wmma_f32_16x16x32_bf16(false, a.v, false, bu1.v, (short)0, cu1, false, false);
    __syncthreads();
  }

  // epilogue: silu(gate) * up -> bf16 act
  int colBase = nt * 64 + wn * 32;
  int mAdd = (lane >= 16) ? 8 : 0;
  #pragma unroll
  for (int c = 0; c < 2; ++c) {
    v8f g = c ? cg1 : cg0;
    v8f u = c ? cu1 : cu0;
    int n = colBase + c * 16 + (lane & 15);
    #pragma unroll
    for (int v = 0; v < 8; ++v) {
      int rloc = mt * 64 + wm * 16 + v + mAdd;
      if (rloc < cnt) {
        float gv = g[v];
        float s = gv / (1.f + __expf(-gv));
        act[(size_t)(offs[e] + rloc) * I_DIM + n] = f2bf(s * u[v]);
      }
    }
  }
}

// ---------------------------------------------------------------- grouped GEMM: down proj
__global__ __launch_bounds__(256) void moe_down_kernel(
    const ushort_t* __restrict__ act, const float* __restrict__ dpw,
    const int* __restrict__ offs, const int* __restrict__ tdn,
    ushort_t* __restrict__ down_b) {
  int tid = blockIdx.x;
  if (tid >= tdn[E_NUM]) return;
  int e = find_expert(tdn, tid);
  int cnt = offs[e + 1] - offs[e];
  int mtiles = (cnt + 63) >> 6;
  int local = tid - tdn[e];
  int mt = local % mtiles;
  int nt = local / mtiles;

  __shared__ __align__(16) ushort_t As[64][32];
  __shared__ __align__(16) ushort_t Bd[64][32];   // transposed [n][k]

  int tx = threadIdx.x;
  // hoisted per-thread addressing; clamp rows (garbage masked at store)
  int ar = tx >> 2, aseg = tx & 3;
  int rr = mt * 64 + ar;
  int srow = (rr < cnt) ? rr : (cnt - 1);
  const ushort_t* aSrc = act + (size_t)(offs[e] + srow) * I_DIM + aseg * 8;
  ushort_t* aDst = &As[ar][aseg * 8];
  int bn = tx & 63, bkh = tx >> 6;
  size_t bBase = ((size_t)e * I_DIM + bkh * 8) * H_DIM + nt * 64 + bn;

  int wid = tx >> 5, lane = tx & 31;
  int wm = wid & 3, wn = wid >> 2;
  v8f c0 = {}, c1 = {};

  for (int kc = 0; kc < I_DIM / 32; ++kc) {
    copy16_g2l(aSrc + kc * 32, aDst);
    {
      const float* dp = dpw + bBase + (size_t)kc * 32 * H_DIM;
      if (kc + 1 < I_DIM / 32)
        __builtin_prefetch(dp + (size_t)32 * H_DIM, 0, 1);
      float w[8];
      #pragma unroll
      for (int i = 0; i < 8; ++i) w[i] = dp[(size_t)i * H_DIM];
      uint4 pw;
      pw.x = pkbf(w[0], w[1]); pw.y = pkbf(w[2], w[3]);
      pw.z = pkbf(w[4], w[5]); pw.w = pkbf(w[6], w[7]);
      *(uint4*)&Bd[bn][bkh * 8] = pw;
    }
    wait_async_copies();
    __syncthreads();

    FragAB a;
    { int row = wm * 16 + (lane & 15); int kb = (lane < 16) ? 0 : 16;
      a.q[0] = *(const uint4*)&As[row][kb];
      a.q[1] = *(const uint4*)&As[row][kb + 8]; }
    FragAB b0, b1;
    { int kb = (lane < 16) ? 0 : 16;
      int cc = wn * 32 + (lane & 15);
      b0.q[0] = *(const uint4*)&Bd[cc][kb];      b0.q[1] = *(const uint4*)&Bd[cc][kb + 8];
      b1.q[0] = *(const uint4*)&Bd[cc + 16][kb]; b1.q[1] = *(const uint4*)&Bd[cc + 16][kb + 8]; }

    c0 = __builtin_amdgcn_wmma_f32_16x16x32_bf16(false, a.v, false, b0.v, (short)0, c0, false, false);
    c1 = __builtin_amdgcn_wmma_f32_16x16x32_bf16(false, a.v, false, b1.v, (short)0, c1, false, false);
    __syncthreads();
  }

  int colBase = nt * 64 + wn * 32;
  int mAdd = (lane >= 16) ? 8 : 0;
  #pragma unroll
  for (int c = 0; c < 2; ++c) {
    v8f d = c ? c1 : c0;
    int n = colBase + c * 16 + (lane & 15);
    #pragma unroll
    for (int v = 0; v < 8; ++v) {
      int rloc = mt * 64 + wm * 16 + v + mAdd;
      if (rloc < cnt)
        down_b[(size_t)(offs[e] + rloc) * H_DIM + n] = f2bf(d[v]);
    }
  }
}

// ---------------------------------------------------------------- weighted combine (deterministic gather)
__global__ __launch_bounds__(256) void combine_kernel(const ushort_t* __restrict__ down_b,
                                                      const float* __restrict__ rw,
                                                      const int* __restrict__ inv_row,
                                                      float* __restrict__ out) {
  int t = blockIdx.x;
  __shared__ float w[K_TOP];
  __shared__ int rows[K_TOP];
  if (threadIdx.x < K_TOP) {
    w[threadIdx.x] = rw[t * K_TOP + threadIdx.x];
    rows[threadIdx.x] = inv_row[t * K_TOP + threadIdx.x];
  }
  __syncthreads();
  for (int h = threadIdx.x; h < H_DIM; h += 256) {
    float acc = 0.f;
    #pragma unroll
    for (int k = 0; k < K_TOP; ++k)
      acc += w[k] * bf2f(down_b[(size_t)rows[k] * H_DIM + h]);
    out[(size_t)t * H_DIM + h] = acc;
  }
}

// ---------------------------------------------------------------- launch
extern "C" void kernel_launch(void* const* d_in, const int* in_sizes, int n_in,
                              void* d_out, int out_size, void* d_ws, size_t ws_size,
                              hipStream_t stream) {
  const float* x   = (const float*)d_in[0];   // [T, H]
  const float* gw  = (const float*)d_in[1];   // [H, E]
  const float* gpw = (const float*)d_in[2];   // [E, H, I]
  const float* upw = (const float*)d_in[3];   // [E, H, I]
  const float* dpw = (const float*)d_in[4];   // [E, I, H]
  float* out = (float*)d_out;

  char* ws = (char*)d_ws;
  size_t off = 0;
  auto alloc = [&](size_t bytes) -> void* {
    void* p = ws + off;
    off += (bytes + 255) & ~(size_t)255;
    return p;
  };
  ushort_t* xb     = (ushort_t*)alloc((size_t)T_TOK * H_DIM * 2);
  float*    rw     = (float*)   alloc((size_t)TK * 4);
  int*      sel    = (int*)     alloc((size_t)TK * 4);
  int*      counts = (int*)     alloc((size_t)E_NUM * 4);
  int*      offs   = (int*)     alloc((size_t)(E_NUM + 1) * 4);
  int*      tgu    = (int*)     alloc((size_t)(E_NUM + 1) * 4);
  int*      tdn    = (int*)     alloc((size_t)(E_NUM + 1) * 4);
  int*      row_tk = (int*)     alloc((size_t)TK * 4);
  int*      invrow = (int*)     alloc((size_t)TK * 4);
  ushort_t* act    = (ushort_t*)alloc((size_t)TK * I_DIM * 2);
  ushort_t* down_b = (ushort_t*)alloc((size_t)TK * H_DIM * 2);

  (void)hipMemsetAsync(counts, 0, E_NUM * sizeof(int), stream);

  cvt_x_kernel<<<(T_TOK * H_DIM / 4 + 255) / 256, 256, 0, stream>>>(x, xb, T_TOK * H_DIM / 4);
  router_kernel<<<T_TOK, 64, 0, stream>>>(x, gw, rw, sel);
  count_kernel<<<(TK + 255) / 256, 256, 0, stream>>>(sel, counts);
  build_kernel<<<1, 64, 0, stream>>>(sel, counts, offs, tgu, tdn, row_tk, invrow);

  // upper bounds on tile counts: sum ceil(cnt/64) <= TK/64 + E
  const int maxGU = (TK / 64 + E_NUM) * (I_DIM / 64);   // 3840
  const int maxDN = (TK / 64 + E_NUM) * (H_DIM / 64);   // 10240
  moe_gateup_kernel<<<maxGU, 256, 0, stream>>>(xb, gpw, upw, offs, tgu, row_tk, act);
  moe_down_kernel<<<maxDN, 256, 0, stream>>>(act, dpw, offs, tdn, down_b);
  combine_kernel<<<T_TOK, 256, 0, stream>>>(down_b, rw, invrow, out);
}
